// PWC_net_10204842295955
// MI455X (gfx1250) — matrix operand order repaired and checked by reading, not compiled, not run on hardware
//
#include <hip/hip_runtime.h>
#include <hip/hip_bf16.h>
#include <math.h>

// ---------------------------------------------------------------------------
// Types for CDNA5 WMMA (wave32): D(f32 16x16) = A(bf16 16x32) * B(bf16 32x16) + C
// ---------------------------------------------------------------------------
typedef __attribute__((ext_vector_type(16))) __bf16 v16bf;
typedef __attribute__((ext_vector_type(8)))  __bf16 v8bf;
typedef __attribute__((ext_vector_type(8)))  float  v8f;

// ---------------------------------------------------------------------------
// Pack fp32 OIHW conv weights into zero-padded tap-major bf16:
//   dst[co][tap*Cinp + ci] = src[co][ci][tap],  Cp = ceil(Cout/32)*32,
//   Cinp = ceil(Cin/32)*32.  Rows 64B aligned; every 32-wide K-step = one tap.
// ---------------------------------------------------------------------------
__global__ void pack_w_kernel(const float* __restrict__ src, __bf16* __restrict__ dst,
                              int Cout, int Cin, int Cinp, int n) {
    int i = blockIdx.x * blockDim.x + threadIdx.x;
    if (i >= n) return;
    int Kp  = 9 * Cinp;
    int co  = i / Kp;
    int r   = i - co * Kp;
    int tap = r / Cinp;
    int ci  = r - tap * Cinp;
    float v = (co < Cout && ci < Cin) ? src[(co * Cin + ci) * 9 + tap] : 0.0f;
    dst[i] = (__bf16)v;
}

__global__ void zero_bf16_kernel(__bf16* __restrict__ p, int n) {
    int i = blockIdx.x * blockDim.x + threadIdx.x;
    if (i < n) p[i] = (__bf16)0.0f;
}

// ---------------------------------------------------------------------------
// Implicit-GEMM 3x3 conv on NHWC bf16 activations, bf16 WMMA, f32 accumulate.
// One wave computes 32(Cout) x 32(pixel): 4 accumulators, each K-step does
// 4 aligned b128 A loads + 4 aligned b128 B loads -> 4 wmma (2:1 load:wmma,
// A and B fragments each reused twice).  Per-tap geometry hoisted (unrolled
// 9x); out-of-bounds taps redirect B pointers to a zero page.  Channels >=
// Cin are annulled by zero-padded A weights.  Epilogue: bias + LeakyReLU,
// 4 b128 stores.
// ---------------------------------------------------------------------------
__global__ void __launch_bounds__(128)
conv_nhwc_wmma_kernel(const __bf16* __restrict__ X, int XCp, int cinoff, int Cin,
                      int Cinp, int Hin, int Win,
                      const __bf16* __restrict__ Wb, const float* __restrict__ bias,
                      __bf16* __restrict__ Y, int YCp, int coutoff, int Cout,
                      int Hout, int Wout, int stride, int pad, int dil, int leaky,
                      int Bn, const __bf16* __restrict__ zp) {
    const int lane = threadIdx.x & 31;
    const int wave = threadIdx.x >> 5;
    const int half = lane >> 4;      // 0: lanes 0-15, 1: lanes 16-31
    const int ln   = lane & 15;

    const int HWo    = Hout * Wout;
    const int Npix   = Bn * HWo;
    const int Ntiles = (Npix + 31) >> 5;   // 32-pixel tiles

    const int nt = blockIdx.x * 4 + wave;  // pixel-tile index
    const int mt = blockIdx.y;             // 32-wide cout-tile index
    if (nt >= Ntiles) return;              // wave-uniform exit

    // Two pixels per lane (N sub-tiles 0 and 1)
    const int  q0  = nt * 32 + ln;
    const int  q1  = q0 + 16;
    const bool qv0 = (q0 < Npix);
    const bool qv1 = (q1 < Npix);
    const int  qc0 = qv0 ? q0 : 0;
    const int  qc1 = qv1 ? q1 : 0;
    const int  n0  = qc0 / HWo,        n1  = qc1 / HWo;
    const int  rp0 = qc0 - n0 * HWo,   rp1 = qc1 - n1 * HWo;
    const int  oy0 = rp0 / Wout,       oy1 = rp1 / Wout;
    const int  ox0 = rp0 - oy0 * Wout, ox1 = rp1 - oy1 * Wout;
    const int  oyb0 = oy0 * stride - pad, oyb1 = oy1 * stride - pad;
    const int  oxb0 = ox0 * stride - pad, oxb1 = ox1 * stride - pad;

    const int Kp = 9 * Cinp;
    const __bf16* Wr0 = Wb + (mt * 32 + ln) * Kp;
    const __bf16* Wr1 = Wr0 + 16 * Kp;
    const int chalf = 16 * half;

    v8f acc00 = {};   // m-tile 0, n-tile 0
    v8f acc10 = {};   // m-tile 1, n-tile 0
    v8f acc01 = {};   // m-tile 0, n-tile 1
    v8f acc11 = {};   // m-tile 1, n-tile 1

#pragma unroll
    for (int tap = 0; tap < 9; ++tap) {
        const int ky = tap / 3, kx = tap - (tap / 3) * 3;   // constants after unroll
        const int iy0 = oyb0 + ky * dil, iy1 = oyb1 + ky * dil;
        const int ix0 = oxb0 + kx * dil, ix1 = oxb1 + kx * dil;
        const bool tv0 = qv0 & (iy0 >= 0) & (iy0 < Hin) & (ix0 >= 0) & (ix0 < Win);
        const bool tv1 = qv1 & (iy1 >= 0) & (iy1 < Hin) & (ix1 >= 0) & (ix1 < Win);
        const int iyc0 = min(max(iy0, 0), Hin - 1), iyc1 = min(max(iy1, 0), Hin - 1);
        const int ixc0 = min(max(ix0, 0), Win - 1), ixc1 = min(max(ix1, 0), Win - 1);
        const __bf16* Xp0 =
            tv0 ? (X + (size_t)((n0 * Hin + iyc0) * Win + ixc0) * XCp + cinoff) : zp;
        const __bf16* Xp1 =
            tv1 ? (X + (size_t)((n1 * Hin + iyc1) * Win + ixc1) * XCp + cinoff) : zp;
        const __bf16* Wt0 = Wr0 + tap * Cinp;
        const __bf16* Wt1 = Wr1 + tap * Cinp;

        for (int cb = 0; cb < Cinp; cb += 32) {
            v8bf a0lo = *(const v8bf*)(Wt0 + cb + half * 8);
            v8bf a0hi = *(const v8bf*)(Wt0 + cb + 16 + half * 8);
            v8bf a1lo = *(const v8bf*)(Wt1 + cb + half * 8);
            v8bf a1hi = *(const v8bf*)(Wt1 + cb + 16 + half * 8);
            v8bf b0lo = *(const v8bf*)(Xp0 + cb + half * 8);
            v8bf b0hi = *(const v8bf*)(Xp0 + cb + 16 + half * 8);
            v8bf b1lo = *(const v8bf*)(Xp1 + cb + half * 8);
            v8bf b1hi = *(const v8bf*)(Xp1 + cb + 16 + half * 8);
            v16bf a0, a1, bf0, bf1;
#pragma unroll
            for (int e = 0; e < 8; ++e) {
                a0[e]  = a0lo[e]; a0[e + 8]  = a0hi[e];
                a1[e]  = a1lo[e]; a1[e + 8]  = a1hi[e];
                bf0[e] = b0lo[e]; bf0[e + 8] = b0hi[e];
                bf1[e] = b1lo[e]; bf1[e + 8] = b1hi[e];
            }
            acc00 = __builtin_amdgcn_wmma_f32_16x16x32_bf16(
                false, a0, false, bf0, (short)0, acc00, false, false);
            acc10 = __builtin_amdgcn_wmma_f32_16x16x32_bf16(
                false, a1, false, bf0, (short)0, acc10, false, false);
            acc01 = __builtin_amdgcn_wmma_f32_16x16x32_bf16(
                false, a0, false, bf1, (short)0, acc01, false, false);
            acc11 = __builtin_amdgcn_wmma_f32_16x16x32_bf16(
                false, a1, false, bf1, (short)0, acc11, false, false);
        }
    }

    // Epilogue: D layout -> VGPR r holds M = r + 8*half, N = lane%16
    const int gbase = mt * 32 + half * 8;
    v8bf o00, o10, o01, o11;
#pragma unroll
    for (int r = 0; r < 8; ++r) {
        int co0 = gbase + r;
        int co1 = co0 + 16;
        float b0 = bias[min(co0, Cout - 1)];
        float b1 = bias[min(co1, Cout - 1)];
        float v00 = acc00[r] + b0, v10 = acc10[r] + b1;
        float v01 = acc01[r] + b0, v11 = acc11[r] + b1;
        if (leaky) {
            v00 = (v00 >= 0.0f) ? v00 : 0.1f * v00;
            v10 = (v10 >= 0.0f) ? v10 : 0.1f * v10;
            v01 = (v01 >= 0.0f) ? v01 : 0.1f * v01;
            v11 = (v11 >= 0.0f) ? v11 : 0.1f * v11;
        }
        o00[r] = (__bf16)v00; o10[r] = (__bf16)v10;
        o01[r] = (__bf16)v01; o11[r] = (__bf16)v11;
    }
    if (qv0) {
        __bf16* Yp0 = Y + (size_t)((n0 * Hout + oy0) * Wout + ox0) * YCp + coutoff + gbase;
        *(v8bf*)(Yp0)      = o00;
        *(v8bf*)(Yp0 + 16) = o10;
    }
    if (qv1) {
        __bf16* Yp1 = Y + (size_t)((n1 * Hout + oy1) * Wout + ox1) * YCp + coutoff + gbase;
        *(v8bf*)(Yp1)      = o01;
        *(v8bf*)(Yp1 + 16) = o11;
    }
}

// ---------------------------------------------------------------------------
// Input adapter: bilinear align-corners upsample, NCHW fp32 -> NHWC bf16,
// zero-filling padded channels.  One thread per output pixel.
// ---------------------------------------------------------------------------
__global__ void upsample_in_kernel(const float* __restrict__ X, __bf16* __restrict__ Y,
                                   int Bn, int Cin, int Hin, int Win,
                                   int Hout, int Wout, int Cp) {
    int i = blockIdx.x * blockDim.x + threadIdx.x;
    int total = Bn * Hout * Wout;
    if (i >= total) return;
    int x = i % Wout; int t = i / Wout;
    int y = t % Hout; int b = t / Hout;

    float sy = (float)y * (float)(Hin - 1) / (float)(Hout - 1);
    float sx = (float)x * (float)(Win - 1) / (float)(Wout - 1);
    int y0 = (int)floorf(sy); if (y0 < 0) y0 = 0; if (y0 > Hin - 1) y0 = Hin - 1;
    int x0 = (int)floorf(sx); if (x0 < 0) x0 = 0; if (x0 > Win - 1) x0 = Win - 1;
    int y1 = y0 + 1 < Hin ? y0 + 1 : Hin - 1;
    int x1 = x0 + 1 < Win ? x0 + 1 : Win - 1;
    float wy = sy - (float)y0, wx = sx - (float)x0;

    __bf16* yp = Y + (size_t)i * Cp;
    for (int c = 0; c < Cp; ++c) {
        float v = 0.0f;
        if (c < Cin) {
            const float* p = X + ((size_t)(b * Cin + c)) * Hin * Win;
            float v00 = p[y0 * Win + x0], v01 = p[y0 * Win + x1];
            float v10 = p[y1 * Win + x0], v11 = p[y1 * Win + x1];
            float r0 = v00 * (1.0f - wy) + v10 * wy;
            float r1 = v01 * (1.0f - wy) + v11 * wy;
            v = r0 * (1.0f - wx) + r1 * wx;
        }
        yp[c] = (__bf16)v;
    }
}

// ---------------------------------------------------------------------------
// Output adapter: bilinear align-corners upsample, NHWC bf16 -> NCHW fp32,
// with linear pre-multiplier.
// ---------------------------------------------------------------------------
__global__ void upsample_out_kernel(const __bf16* __restrict__ X, int XCp,
                                    float* __restrict__ Y,
                                    int Bn, int C, int Hin, int Win,
                                    int Hout, int Wout, float premul) {
    int i = blockIdx.x * blockDim.x + threadIdx.x;
    int total = Bn * C * Hout * Wout;
    if (i >= total) return;
    int x = i % Wout; int t = i / Wout;
    int y = t % Hout; t /= Hout;
    int c = t % C;    int b = t / C;

    float sy = (float)y * (float)(Hin - 1) / (float)(Hout - 1);
    float sx = (float)x * (float)(Win - 1) / (float)(Wout - 1);
    int y0 = (int)floorf(sy); if (y0 < 0) y0 = 0; if (y0 > Hin - 1) y0 = Hin - 1;
    int x0 = (int)floorf(sx); if (x0 < 0) x0 = 0; if (x0 > Win - 1) x0 = Win - 1;
    int y1 = y0 + 1 < Hin ? y0 + 1 : Hin - 1;
    int x1 = x0 + 1 < Win ? x0 + 1 : Win - 1;
    float wy = sy - (float)y0, wx = sx - (float)x0;

    float v00 = (float)X[(size_t)((b * Hin + y0) * Win + x0) * XCp + c];
    float v01 = (float)X[(size_t)((b * Hin + y0) * Win + x1) * XCp + c];
    float v10 = (float)X[(size_t)((b * Hin + y1) * Win + x0) * XCp + c];
    float v11 = (float)X[(size_t)((b * Hin + y1) * Win + x1) * XCp + c];
    float r0 = v00 * (1.0f - wy) + v10 * wy;
    float r1 = v01 * (1.0f - wy) + v11 * wy;
    Y[i] = premul * (r0 * (1.0f - wx) + r1 * wx);
}

// ---------------------------------------------------------------------------
// 81-displacement cost volume on NHWC bf16, mean over channels, fused
// LeakyReLU(0.1); unit-stride channel loop.  Writes a channel slice of E.
// ---------------------------------------------------------------------------
__global__ void corr_nhwc_kernel(const __bf16* __restrict__ A, const __bf16* __restrict__ Bm,
                                 int C, int Cp, __bf16* __restrict__ Y, int YCp, int yoff,
                                 int Bn, int H, int W) {
    int i = blockIdx.x * blockDim.x + threadIdx.x;
    int total = Bn * H * W * 81;
    if (i >= total) return;
    int d = i % 81; int t = i / 81;
    int x = t % W;  t /= W;
    int y = t % H;  int b = t / H;
    int dy = d / 9 - 4, dx = d % 9 - 4;
    int yy = y + dy, xx = x + dx;
    float s = 0.0f;
    if (yy >= 0 && yy < H && xx >= 0 && xx < W) {
        const __bf16* pa = A  + (size_t)((b * H + y)  * W + x)  * Cp;
        const __bf16* pb = Bm + (size_t)((b * H + yy) * W + xx) * Cp;
        for (int c = 0; c < C; ++c) s += (float)pa[c] * (float)pb[c];
    }
    s /= (float)C;
    s = (s >= 0.0f) ? s : 0.1f * s;
    Y[(size_t)((b * H + y) * W + x) * YCp + yoff + d] = (__bf16)s;
}

// ---------------------------------------------------------------------------
// Backward bilinear warp on NHWC bf16; invalid corners read the zero page.
// One thread per pixel, unit-stride channel loop.
// ---------------------------------------------------------------------------
__global__ void warp_nhwc_kernel(const __bf16* __restrict__ X, int XCp, int C,
                                 const __bf16* __restrict__ F, int FCp,
                                 __bf16* __restrict__ Y, int YCp,
                                 int Bn, int H, int W, float fs,
                                 const __bf16* __restrict__ zp) {
    int i = blockIdx.x * blockDim.x + threadIdx.x;
    int total = Bn * H * W;
    if (i >= total) return;
    int x = i % W; int t = i / W;
    int y = t % H; int b = t / H;

    const __bf16* Fp = F + (size_t)i * FCp;
    float px = (float)x + (float)Fp[0] * fs;
    float py = (float)y + (float)Fp[1] * fs;
    float x0f = floorf(px), y0f = floorf(py);
    int x0 = (int)x0f, y0 = (int)y0f;
    float wx = px - x0f, wy = py - y0f;

    auto corner = [&](int yi, int xi) -> const __bf16* {
        bool v = (xi >= 0) && (xi < W) && (yi >= 0) && (yi < H);
        return v ? (X + (size_t)((b * H + yi) * W + xi) * XCp) : zp;
    };
    const __bf16* p00 = corner(y0, x0);
    const __bf16* p01 = corner(y0, x0 + 1);
    const __bf16* p10 = corner(y0 + 1, x0);
    const __bf16* p11 = corner(y0 + 1, x0 + 1);

    __bf16* yp = Y + (size_t)i * YCp;
    float w00 = (1.f - wx) * (1.f - wy), w01 = wx * (1.f - wy);
    float w10 = (1.f - wx) * wy,         w11 = wx * wy;
    for (int c = 0; c < C; ++c) {
        float v = (float)p00[c] * w00 + (float)p01[c] * w01
                + (float)p10[c] * w10 + (float)p11[c] * w11;
        yp[c] = (__bf16)v;
    }
}

// ---------------------------------------------------------------------------
// ConvTranspose2d(k=4, s=2, p=1) on NHWC bf16; torch weights (Cin, Cout, 4, 4).
// ---------------------------------------------------------------------------
__global__ void deconv_nhwc_kernel(const __bf16* __restrict__ X, int XCp, int Cin,
                                   int Hin, int Win,
                                   const float* __restrict__ Wt,
                                   const float* __restrict__ bias,
                                   __bf16* __restrict__ Y, int YCp, int Cout, int Bn) {
    int Hout = 2 * Hin, Wout = 2 * Win;
    int i = blockIdx.x * blockDim.x + threadIdx.x;
    int total = Bn * Hout * Wout * Cout;
    if (i >= total) return;
    int co = i % Cout; int t = i / Cout;
    int ox = t % Wout; t /= Wout;
    int oy = t % Hout; int b = t / Hout;

    float acc = bias[co];
    for (int ky = 0; ky < 4; ++ky) {
        int ty = oy + 1 - ky;
        if (ty < 0 || (ty & 1)) continue;
        int iy = ty >> 1; if (iy >= Hin) continue;
        for (int kx = 0; kx < 4; ++kx) {
            int tx = ox + 1 - kx;
            if (tx < 0 || (tx & 1)) continue;
            int ix = tx >> 1; if (ix >= Win) continue;
            const __bf16* xp = X + (size_t)((b * Hin + iy) * Win + ix) * XCp;
            const float*  wp = Wt + (co * 4 + ky) * 4 + kx;
            for (int ci = 0; ci < Cin; ++ci)
                acc += wp[ci * Cout * 16] * (float)xp[ci];
        }
    }
    Y[(size_t)((b * Hout + oy) * Wout + ox) * YCp + co] = (__bf16)acc;
}

// ---------------------------------------------------------------------------
// Copy C channels between channel slices of NHWC bf16 buffers.
// ---------------------------------------------------------------------------
__global__ void copy_slice_nhwc_kernel(const __bf16* __restrict__ X, int XCp, int xoff,
                                       __bf16* __restrict__ Y, int YCp, int yoff,
                                       int npix, int C) {
    int i = blockIdx.x * blockDim.x + threadIdx.x;
    int total = npix * C;
    if (i >= total) return;
    int c = i % C, p = i / C;
    Y[(size_t)p * YCp + yoff + c] = X[(size_t)p * XCp + xoff + c];
}

__global__ void add_nhwc_kernel(const __bf16* __restrict__ A, const __bf16* __restrict__ B,
                                __bf16* __restrict__ Y, int Cp, int C, int npix) {
    int i = blockIdx.x * blockDim.x + threadIdx.x;
    int total = npix * C;
    if (i >= total) return;
    int c = i % C, p = i / C;
    Y[(size_t)p * Cp + c] = (__bf16)((float)A[(size_t)p * Cp + c] +
                                     (float)B[(size_t)p * Cp + c]);
}

// ---------------------------------------------------------------------------
// Host orchestration
// ---------------------------------------------------------------------------
static inline int cdiv_i(int a, int b) { return (a + b - 1) / b; }

extern "C" void kernel_launch(void* const* d_in, const int* in_sizes, int n_in,
                              void* d_out, int out_size, void* d_ws, size_t ws_size,
                              hipStream_t stream) {
    (void)in_sizes; (void)n_in; (void)out_size; (void)ws_size;
    const int Bn = 2;
    const float* inA = (const float*)d_in[0];
    const float* inB = (const float*)d_in[1];
    float* out = (float*)d_out;

    char* base = (char*)d_ws;
    size_t wo = 0;
    auto allocB = [&](size_t bytes) -> void* {
        void* p = base + wo;
        wo = (wo + bytes + 255) & ~(size_t)255;
        return p;
    };
    auto aBF = [&](size_t n) { return (__bf16*)allocB(n * sizeof(__bf16) + 128); };
    auto cp32 = [&](int c) { return cdiv_i(c, 32) * 32; };

    // ---- Unpack params in setup_inputs() dict order
    struct PW { const float* w; const float* b; };
    int pi = 2;
    auto nx = [&]() { PW p; p.w = (const float*)d_in[pi++]; p.b = (const float*)d_in[pi++]; return p; };

    PW featP[6][3];
    for (int i = 0; i < 6; ++i) for (int j = 0; j < 3; ++j) featP[i][j] = nx();
    PW estP[5][6];   // est5, est4, est3, est2, est1
    for (int i = 0; i < 5; ++i) for (int j = 0; j < 6; ++j) estP[i][j] = nx();
    PW upP[4][2];    // up5..up2, each {up_flow, up_feat}
    for (int i = 0; i < 4; ++i) for (int j = 0; j < 2; ++j) upP[i][j] = nx();
    PW ctxP[7];
    for (int i = 0; i < 7; ++i) ctxP[i] = nx();

    // ---- Zero page for out-of-bounds B rows / warp corners
    __bf16* zp = aBF(1024);
    zero_bf16_kernel<<<4, 256, 0, stream>>>(zp, 1024);

    // ---- Pack all 3x3 conv weights to padded tap-major bf16
    auto packW = [&](const float* w, int Cout, int Cin) -> __bf16* {
        int Cinp = cp32(Cin), Cp = cp32(Cout);
        int n = Cp * 9 * Cinp;
        __bf16* p = aBF((size_t)n);
        pack_w_kernel<<<cdiv_i(n, 256), 256, 0, stream>>>(w, p, Cout, Cin, Cinp, n);
        return p;
    };

    const int fco[6] = {16, 32, 64, 96, 128, 196};
    const int fci[6] = {3, 16, 32, 64, 96, 128};
    __bf16* featW[6][3];
    for (int i = 0; i < 6; ++i) {
        featW[i][0] = packW(featP[i][0].w, fco[i], fci[i]);
        featW[i][1] = packW(featP[i][1].w, fco[i], fco[i]);
        featW[i][2] = packW(featP[i][2].w, fco[i], fco[i]);
    }
    const int efin[5]  = {81, 213, 181, 149, 117};
    const int ecout[6] = {128, 128, 96, 64, 32, 2};
    const int ecinb[6] = {0, 128, 256, 352, 416, 448};
    __bf16* estW[5][6];
    for (int l = 0; l < 5; ++l)
        for (int j = 0; j < 6; ++j)
            estW[l][j] = packW(estP[l][j].w, ecout[j], ecinb[j] + efin[l]);
    const int ctxco[7] = {128, 128, 128, 96, 64, 32, 2};
    const int ctxci[7] = {565, 128, 128, 128, 96, 64, 32};
    __bf16* ctxW[7];
    for (int i = 0; i < 7; ++i) ctxW[i] = packW(ctxP[i].w, ctxco[i], ctxci[i]);

    // ---- Conv launcher (NHWC bf16 in/out); one wave = 32 cout x 32 pixels
    auto conv = [&](const __bf16* X, int XCp, int cinoff, int Cin, int Hin, int Win,
                    const __bf16* Wb, const float* bias,
                    __bf16* Y, int YCp, int coutoff, int Cout, int Hout, int Wout,
                    int stride, int pad, int dil, int leaky) {
        int Cinp = cp32(Cin);
        int Npix = Bn * Hout * Wout;
        int Nt = cdiv_i(Npix, 32), Mt = cdiv_i(Cout, 32);
        dim3 g(cdiv_i(Nt, 4), Mt);
        conv_nhwc_wmma_kernel<<<g, 128, 0, stream>>>(
            X, XCp, cinoff, Cin, Cinp, Hin, Win, Wb, bias,
            Y, YCp, coutoff, Cout, Hout, Wout, stride, pad, dil, leaky, Bn, zp);
    };
    auto deconv = [&](const __bf16* X, int XCp, int Cin, int Hin, int Win, PW p,
                      __bf16* Y, int YCp) {
        int n = Bn * (2 * Hin) * (2 * Win) * 2;
        deconv_nhwc_kernel<<<cdiv_i(n, 256), 256, 0, stream>>>(
            X, XCp, Cin, Hin, Win, p.w, p.b, Y, YCp, 2, Bn);
    };

    // ---- 1) Input adapters: NCHW f32 (384) -> NHWC bf16 (768), Cp=32
    __bf16* upA = aBF((size_t)Bn * 768 * 768 * 32);
    __bf16* upB = aBF((size_t)Bn * 768 * 768 * 32);
    {
        int n = Bn * 768 * 768;
        upsample_in_kernel<<<cdiv_i(n, 256), 256, 0, stream>>>(inA, upA, Bn, 3, 384, 384, 768, 768, 32);
        upsample_in_kernel<<<cdiv_i(n, 256), 256, 0, stream>>>(inB, upB, Bn, 3, 384, 384, 768, 768, 32);
    }

    // ---- 2) Feature pyramid
    const int fH[6]  = {384, 192, 96, 48, 24, 12};
    int fCp[6]; for (int i = 0; i < 6; ++i) fCp[i] = cp32(fco[i]);
    __bf16* s1 = aBF((size_t)Bn * 384 * 384 * 32);
    __bf16* s2 = aBF((size_t)Bn * 384 * 384 * 32);
    __bf16* b1 = aBF((size_t)Bn * 384 * 384 * 32);
    __bf16* feat[2][5];   // [side][f1..f5]
    for (int sd = 0; sd < 2; ++sd) {
        const __bf16* cur = sd ? upB : upA;
        int curC = 3, curCp = 32, curH = 768;
        for (int bk = 0; bk < 6; ++bk) {
            int Ho = fH[bk], Co = fco[bk], Cp = fCp[bk];
            conv(cur, curCp, 0, curC, curH, curH, featW[bk][0], featP[bk][0].b,
                 s1, Cp, 0, Co, Ho, Ho, 2, 1, 1, 1);
            conv(s1, Cp, 0, Co, Ho, Ho, featW[bk][1], featP[bk][1].b,
                 s2, Cp, 0, Co, Ho, Ho, 1, 1, 1, 1);
            __bf16* dst;
            if (bk == 0) dst = b1;
            else { dst = aBF((size_t)Bn * Ho * Ho * Cp); feat[sd][bk - 1] = dst; }
            conv(s2, Cp, 0, Co, Ho, Ho, featW[bk][2], featP[bk][2].b,
                 dst, Cp, 0, Co, Ho, Ho, 1, 1, 1, 1);
            cur = dst; curC = Co; curCp = Cp; curH = Ho;
        }
    }

    // ---- Flow estimator on dense-concat E (Ct = 448+fin, ECp = cp32(Ct)).
    // Slices: x0@448, c1@320, c2@192, c3@96, c4@32, c5@0.
    auto flowEst = [&](__bf16* E, int ECp, int l, int H, __bf16* flow) {
        int fin = efin[l];
        const int inoff[6]  = {448, 320, 192, 96, 32, 0};
        const int outoff[5] = {320, 192, 96, 32, 0};
        for (int j = 0; j < 5; ++j)
            conv(E, ECp, inoff[j], ecinb[j] + fin, H, H, estW[l][j], estP[l][j].b,
                 E, ECp, outoff[j], ecout[j], H, H, 1, 1, 1, 1);
        conv(E, ECp, 0, 448 + fin, H, H, estW[l][5], estP[l][5].b,
             flow, 32, 0, 2, H, H, 1, 1, 1, 0);
    };

    // ---- 3) Level 5 (12x12)
    int E5Cp = cp32(529);
    __bf16* E5 = aBF((size_t)Bn * 12 * 12 * E5Cp);
    {
        int n = Bn * 12 * 12 * 81;
        corr_nhwc_kernel<<<cdiv_i(n, 256), 256, 0, stream>>>(
            feat[0][4], feat[1][4], 196, fCp[5], E5, E5Cp, 448, Bn, 12, 12);
    }
    __bf16* flow5 = aBF((size_t)Bn * 12 * 12 * 32);
    flowEst(E5, E5Cp, 0, 12, flow5);
    __bf16* uf = aBF((size_t)Bn * 24 * 24 * 32);
    __bf16* uT = aBF((size_t)Bn * 24 * 24 * 32);
    deconv(flow5, 32, 2, 12, 12, upP[0][0], uf, 32);
    deconv(E5, E5Cp, 529, 12, 12, upP[0][1], uT, 32);

    // ---- 4) Levels 4..1
    struct LL { int fi; int C; int H; float fs; int est; int up; };
    const LL Ls[4] = {
        {3, 128, 24, 0.625f, 1, 1},
        {2, 96,  48, 1.25f,  2, 2},
        {1, 64,  96, 2.5f,   3, 3},
        {0, 32, 192, 5.0f,   4, -1},
    };
    __bf16* flow1 = nullptr;
    __bf16* E1 = nullptr;
    int E1Cp = 0;
    for (int li = 0; li < 4; ++li) {
        const LL L = Ls[li];
        int H = L.H, C = L.C, Cp = cp32(C);
        int fin = 81 + C + 4, Ct = 448 + fin, ECp = cp32(Ct);
        int npix = Bn * H * H;
        __bf16* wb = aBF((size_t)npix * Cp);
        warp_nhwc_kernel<<<cdiv_i(npix, 256), 256, 0, stream>>>(
            feat[1][L.fi], Cp, C, uf, 32, wb, Cp, Bn, H, H, L.fs, zp);
        __bf16* E = aBF((size_t)npix * ECp);
        {
            int n = npix * 81;
            corr_nhwc_kernel<<<cdiv_i(n, 256), 256, 0, stream>>>(
                feat[0][L.fi], wb, C, Cp, E, ECp, 448, Bn, H, H);
        }
        copy_slice_nhwc_kernel<<<cdiv_i(npix * C, 256), 256, 0, stream>>>(
            feat[0][L.fi], Cp, 0, E, ECp, 448 + 81, npix, C);
        copy_slice_nhwc_kernel<<<cdiv_i(npix * 2, 256), 256, 0, stream>>>(
            uf, 32, 0, E, ECp, 448 + 81 + C, npix, 2);
        copy_slice_nhwc_kernel<<<cdiv_i(npix * 2, 256), 256, 0, stream>>>(
            uT, 32, 0, E, ECp, 448 + 81 + C + 2, npix, 2);
        __bf16* fl = aBF((size_t)npix * 32);
        flowEst(E, ECp, L.est, H, fl);
        if (L.up >= 0) {
            __bf16* nuf = aBF((size_t)Bn * (2 * H) * (2 * H) * 32);
            __bf16* nuT = aBF((size_t)Bn * (2 * H) * (2 * H) * 32);
            deconv(fl, 32, 2, H, H, upP[L.up][0], nuf, 32);
            deconv(E, ECp, Ct, H, H, upP[L.up][1], nuT, 32);
            uf = nuf; uT = nuT;
        } else {
            flow1 = fl; E1 = E; E1Cp = ECp;
        }
    }

    // ---- 5) Context net (dilated convs) + residue, at 192x192
    __bf16* t1 = aBF((size_t)Bn * 192 * 192 * 128);
    __bf16* t2 = aBF((size_t)Bn * 192 * 192 * 128);
    const int cdil[6] = {1, 2, 4, 8, 16, 1};
    const __bf16* cur = E1;
    int curCp = E1Cp;
    __bf16* dst = t1;
    for (int i = 0; i < 6; ++i) {
        conv(cur, curCp, 0, ctxci[i], 192, 192, ctxW[i], ctxP[i].b,
             dst, 128, 0, ctxco[i], 192, 192, 1, cdil[i], cdil[i], 1);
        cur = dst; curCp = 128;
        dst = (dst == t1) ? t2 : t1;
    }
    __bf16* res = aBF((size_t)Bn * 192 * 192 * 32);
    conv(cur, curCp, 0, 32, 192, 192, ctxW[6], ctxP[6].b,
         res, 32, 0, 2, 192, 192, 1, 1, 1, 0);

    // ---- 6) pred_flow = flow1 + residue; scale by 10; 2x upsample -> NCHW f32
    __bf16* pf = aBF((size_t)Bn * 192 * 192 * 32);
    {
        int npix = Bn * 192 * 192;
        add_nhwc_kernel<<<cdiv_i(npix * 2, 256), 256, 0, stream>>>(flow1, res, pf, 32, 2, npix);
        int no = Bn * 2 * 384 * 384;
        upsample_out_kernel<<<cdiv_i(no, 256), 256, 0, stream>>>(
            pf, 32, out, Bn, 2, 192, 192, 384, 384, 10.0f);
    }
}
